// MultiQueryAttentionLayerWithDownSampling_1511828488692
// MI455X (gfx1250) — matrix operand order repaired and checked by reading, not compiled, not run on hardware
//
#include <hip/hip_runtime.h>
#include <hip/hip_bf16.h>
#include <math.h>
#include <stdint.h>

// ---------------------------------------------------------------------------
// MI455X (gfx1250) implementation: bf16 WMMA GEMMs + flash attention.
// wave32, v_wmma_f32_16x16x32_bf16, async global->LDS tile staging (ASYNCcnt).
// ---------------------------------------------------------------------------

typedef __attribute__((ext_vector_type(16))) __bf16 v16bf;
typedef __attribute__((ext_vector_type(8)))  float  v8f;

struct __attribute__((aligned(16))) U4 { unsigned x, y, z, w; };
union AB  { v16bf v; U4 q[2]; };            // 32B: one WMMA A/B fragment per lane
union B8  { U4 q; __bf16 h[8]; };           // 8 packed bf16

__device__ __forceinline__ __bf16 f2bf(float f) {
  unsigned u = __builtin_bit_cast(unsigned, f);
  u += 0x7fffu + ((u >> 16) & 1u);          // round-to-nearest-even
  unsigned short s = (unsigned short)(u >> 16);
  return __builtin_bit_cast(__bf16, s);
}

__device__ __forceinline__ v8f wmma_bf16(const AB& a, const AB& b, v8f c) {
  // v_wmma_f32_16x16x32_bf16  D = A*B + C
  return __builtin_amdgcn_wmma_f32_16x16x32_bf16(
      false, a.v, false, b.v, (short)0, c, false, false);
}

// CDNA5 async global->LDS copy: 16B per lane, tracked with ASYNCcnt.
// VDST = LDS byte address (low 32 bits of the generic shared pointer),
// VADDR = 64-bit global address.
__device__ __forceinline__ void async_b128(void* lds, const void* g) {
  asm volatile("global_load_async_to_lds_b128 %0, %1, off"
               :: "v"((unsigned)(uintptr_t)lds),
                  "v"((unsigned long long)(uintptr_t)g)
               : "memory");
}
__device__ __forceinline__ void wait_async0() {
  asm volatile("s_wait_asynccnt 0x0" ::: "memory");
}

// Problem constants
#define B_  16
#define C_  256
#define P_  4096   // 64x64 pixels
#define L_  1024   // 32x32 pixels after stride-2
#define D_  64     // KEY_DIM
#define NH_ 8
#define OC_ 512    // NUM_HEADS*KEY_DIM

// ---------------------------------------------------------------------------
// 1. f32 -> bf16 elementwise convert
// ---------------------------------------------------------------------------
__global__ __launch_bounds__(256) void cvt_bf16_kernel(const float* __restrict__ in,
                                                       __bf16* __restrict__ out,
                                                       long n) {
  long i = (long)blockIdx.x * blockDim.x + threadIdx.x;
  if (i < n) out[i] = f2bf(in[i]);
}

// ---------------------------------------------------------------------------
// 2. Fold BatchNorm into projection weights:
//    Weff[o][c] = W[o][c] * gamma[c]/sqrt(var[c]+eps)
//    bias[o]    = sum_c W[o][c] * (beta[c] - mean[c]*scale[c])
// ---------------------------------------------------------------------------
__global__ __launch_bounds__(256) void fold_kv_kernel(const float* __restrict__ W,
                                                      const float* __restrict__ gamma,
                                                      const float* __restrict__ beta,
                                                      const float* __restrict__ mean,
                                                      const float* __restrict__ var,
                                                      __bf16* __restrict__ Weff,
                                                      float* __restrict__ bias) {
  __shared__ float red[256];
  int o = blockIdx.x, c = threadIdx.x;
  float sc = gamma[c] * rsqrtf(var[c] + 1e-5f);
  float w  = W[o * C_ + c];
  Weff[o * C_ + c] = f2bf(w * sc);
  red[c] = w * (beta[c] - mean[c] * sc);
  __syncthreads();
  for (int s = 128; s > 0; s >>= 1) {
    if (c < s) red[c] += red[c + s];
    __syncthreads();
  }
  if (c == 0) bias[o] = red[0];
}

// ---------------------------------------------------------------------------
// 3. Depthwise 3x3 stride-2 pad-1 conv (k and v paths in one pass over x).
// ---------------------------------------------------------------------------
__global__ __launch_bounds__(256) void dwconv_kernel(const float* __restrict__ x,
                                                     const float* __restrict__ kw,
                                                     const float* __restrict__ vw,
                                                     __bf16* __restrict__ yk,
                                                     __bf16* __restrict__ yv) {
  int l = blockIdx.x * 256 + threadIdx.x;   // 0..1023
  int c = blockIdx.y, b = blockIdx.z;
  int oh = l >> 5, ow = l & 31;
  int ih0 = oh * 2 - 1, iw0 = ow * 2 - 1;
  const float* xp = x + ((long)(b * C_ + c)) * P_;
  const float* kwc = kw + c * 9;
  const float* vwc = vw + c * 9;
  float sk = 0.f, sv = 0.f;
#pragma unroll
  for (int r = 0; r < 3; ++r) {
#pragma unroll
    for (int s = 0; s < 3; ++s) {
      int ih = ih0 + r, iw = iw0 + s;
      if (ih >= 0 && ih < 64 && iw >= 0 && iw < 64) {
        float xv = xp[ih * 64 + iw];
        sk += xv * kwc[r * 3 + s];
        sv += xv * vwc[r * 3 + s];
      }
    }
  }
  long o = ((long)(b * C_ + c)) * L_ + l;
  yk[o] = f2bf(sk);
  yv[o] = f2bf(sv);
}

// ---------------------------------------------------------------------------
// 4. Generic bf16 WMMA GEMM:  C[b] = A (MxK, shared) * Bmat[b] (KxN)
//    Block = 256 threads (8 waves). Block tile = 128(M) x 64(N), K step 64.
//    A tile staged via async global->LDS b128 (full tiles); B tile transposed
//    into LDS via ds stores. 8 v_wmma per wave per stage.
//    Epilogue modes:
//      0: Q  -> scale by 1/sqrt(8), scrambled write to Q[b][h][qq][d] bf16
//      1: K  -> +bias, write Kbf[b][l][d] bf16 (transposed: async-friendly)
//      2: V  -> +bias, scrambled write Vbf[b][d][l] bf16
//      3: O  -> f32 write out[b][m][n]
// ---------------------------------------------------------------------------
__global__ __launch_bounds__(256) void gemm_bf16_kernel(const __bf16* __restrict__ A,
                                                        const __bf16* __restrict__ Bmat,
                                                        const float* __restrict__ bias,
                                                        void* __restrict__ Cout,
                                                        int M, int N, int Kd, int mode) {
  __shared__ __align__(16) __bf16 As[128][64];   // [m][k]  16 KB
  __shared__ __align__(16) __bf16 Bs[64][64];    // [n][k]  8 KB (B transposed)

  int bm0 = blockIdx.x * 128, bn0 = blockIdx.y * 64, b = blockIdx.z;
  int t = threadIdx.x;
  int wave = t >> 5, lane = t & 31, hf = lane >> 4, ln = lane & 15;
  const __bf16* Bb = Bmat + (long)b * Kd * N;
  bool fullM = (bm0 + 128 <= M);

  v8f acc[4] = {};

  for (int k0 = 0; k0 < Kd; k0 += 64) {
    __syncthreads();
    // ---- A tile: 128x64 bf16 = 1024 x 16B slots, 4 per thread ----
    if (fullM) {
#pragma unroll
      for (int i = 0; i < 4; ++i) {
        int s = t + i * 256;
        int row = s >> 3, c8 = s & 7;
        async_b128(&As[row][c8 * 8], A + (long)(bm0 + row) * Kd + k0 + c8 * 8);
      }
    } else {
#pragma unroll
      for (int i = 0; i < 4; ++i) {
        int s = t + i * 256;
        int row = s >> 3, c8 = s & 7;
        U4 val = {0u, 0u, 0u, 0u};
        int gm = bm0 + row;
        if (gm < M) val = *(const U4*)(A + (long)gm * Kd + k0 + c8 * 8);
        *(U4*)&As[row][c8 * 8] = val;
      }
    }
    // ---- B tile 64(k) x 64(n), transposed into Bs[n][k] ----
#pragma unroll
    for (int i = 0; i < 2; ++i) {
      int s = t + i * 256;
      int kk = s & 63, ng = s >> 6;     // kk 0..63, ng 0..7
      B8 bv;
      bv.q = *(const U4*)(Bb + (long)(k0 + kk) * N + bn0 + ng * 8);
#pragma unroll
      for (int j = 0; j < 8; ++j) Bs[ng * 8 + j][kk] = bv.h[j];
    }
    wait_async0();
    __syncthreads();

    int mrow = wave * 16 + ln;
#pragma unroll
    for (int kc = 0; kc < 2; ++kc) {
      AB af, bf_;
      // A-frag: K = (e/8)*16 + hf*8 + e%8 within the 32-wide chunk
      af.q[0] = *(const U4*)&As[mrow][kc * 32 + hf * 8];
      af.q[1] = *(const U4*)&As[mrow][kc * 32 + 16 + hf * 8];
#pragma unroll
      for (int nt = 0; nt < 4; ++nt) {
        int nr = nt * 16 + ln;
        // B-frag: K = (e/8)*16 + hf*8 + e%8 within the 32-wide chunk
        bf_.q[0] = *(const U4*)&Bs[nr][kc * 32 + hf * 8];
        bf_.q[1] = *(const U4*)&Bs[nr][kc * 32 + 16 + hf * 8];
        acc[nt] = wmma_bf16(af, bf_, acc[nt]);
      }
    }
  }

  // Epilogue. C-tile: lane holds rows m = hf*8 + r, col n = ln.
#pragma unroll
  for (int nt = 0; nt < 4; ++nt) {
#pragma unroll
    for (int r = 0; r < 8; ++r) {
      int m = bm0 + wave * 16 + hf * 8 + r;
      int n = bn0 + nt * 16 + ln;
      if (m >= M) continue;
      float val = acc[nt][r];
      if (mode == 0) {
        // q: fold softmax scale; torch .view scramble:
        // h=m/64, cl=m%64, pr=n/64, pc=n%64 -> Q[b][h][cl*64+pr][pc]
        float sv = val * 0.35355339059327373f;  // 1/sqrt(8)
        int hh = m >> 6, cl = m & 63, pr = n >> 6, pc = n & 63;
        long qq = cl * 64 + pr;
        ((__bf16*)Cout)[(((long)b * NH_ + hh) * (long)P_ + qq) * D_ + pc] = f2bf(sv);
      } else if (mode == 1) {
        val += bias[m];
        // K stored [b][l][d] so flash can async-copy contiguous rows
        ((__bf16*)Cout)[((long)b * L_ + n) * D_ + m] = f2bf(val);
      } else if (mode == 2) {
        val += bias[m];
        // v scramble: l = m*16 + n/64, d = n%64; store d-major [b][d][l]
        int l = m * 16 + (n >> 6);
        int d = n & 63;
        ((__bf16*)Cout)[((long)b * D_ + d) * L_ + l] = f2bf(val);
      } else {
        ((float*)Cout)[((long)b * M + m) * (long)N + n] = val;
      }
    }
  }
}

// ---------------------------------------------------------------------------
// 5. Flash attention. grid = (32 qblocks, 8 heads, 16 batch), block = 256.
//    Each wave owns 16 query rows (block = 128 rows); loop L in chunks of 32.
//    K/V tiles double-buffered via async global->LDS b128 copies (ASYNCcnt):
//    next tile streams in while the current tile is consumed by WMMA.
//    Multi-query: K/V shared across the 8 heads.
// ---------------------------------------------------------------------------
__global__ __launch_bounds__(256) void flash_kernel(const __bf16* __restrict__ Qbf,
                                                    const __bf16* __restrict__ Kbf,
                                                    const __bf16* __restrict__ Vbf,
                                                    __bf16* __restrict__ CTX) {
  __shared__ __align__(16) __bf16 KT[2][32][64];   // [buf][l][d]  (B^T for Q*K)
  __shared__ __align__(16) __bf16 VT[2][64][32];   // [buf][d][l]  (B^T for P*V)
  __shared__ __align__(16) __bf16 PS[8][16][32];   // per-wave P staging [m][k]

  int qb = blockIdx.x, h = blockIdx.y, b = blockIdx.z;
  int t = threadIdx.x;
  int wave = t >> 5, lane = t & 31, hf = lane >> 4, ln = lane & 15;

  // Q A-fragments for this wave's 16 rows, kept in registers (2 k-chunks).
  int q0 = qb * 128 + wave * 16;
  const __bf16* Qp = Qbf + (((long)(b * NH_ + h)) * P_ + q0 + ln) * D_;
  AB qf[2];
  qf[0].q[0] = *(const U4*)(Qp + hf * 8);
  qf[0].q[1] = *(const U4*)(Qp + 16 + hf * 8);
  qf[1].q[0] = *(const U4*)(Qp + 32 + hf * 8);
  qf[1].q[1] = *(const U4*)(Qp + 48 + hf * 8);

  v8f acc[4] = {};
  float mi[8], li[8];
#pragma unroll
  for (int r = 0; r < 8; ++r) { mi[r] = -1e30f; li[r] = 0.f; }

  const __bf16* Kb = Kbf + (long)b * L_ * D_;   // [l][d]
  const __bf16* Vb = Vbf + (long)b * D_ * L_;   // [d][l]

  int krow = t >> 3, kseg = t & 7;   // K tile: 32 rows x 8 x 16B
  int vrow = t >> 2, vseg = t & 3;   // V tile: 64 rows x 4 x 16B

  // Prologue: stream tile 0 into buffer 0.
  async_b128(&KT[0][krow][kseg * 8], Kb + (long)krow * D_ + kseg * 8);
  async_b128(&VT[0][vrow][vseg * 8], Vb + (long)vrow * L_ + vseg * 8);
  wait_async0();
  __syncthreads();

  int cur = 0;
  for (int l0 = 0; l0 < L_; l0 += 32) {
    int nxt = cur ^ 1;
    if (l0 + 32 < L_) {   // stream next tile while computing on current
      async_b128(&KT[nxt][krow][kseg * 8], Kb + (long)(l0 + 32 + krow) * D_ + kseg * 8);
      async_b128(&VT[nxt][vrow][vseg * 8], Vb + (long)vrow * L_ + l0 + 32 + vseg * 8);
    }

    // S = Q * K^T  (16 rows x 32 keys) -> two 16x16 C-tiles
    v8f s[2] = {};
    AB kf;
#pragma unroll
    for (int nt = 0; nt < 2; ++nt) {
      int nr = nt * 16 + ln;
      kf.q[0] = *(const U4*)&KT[cur][nr][hf * 8];           // d-chunk 0 (K=0..31)
      kf.q[1] = *(const U4*)&KT[cur][nr][16 + hf * 8];
      s[nt] = wmma_bf16(qf[0], kf, s[nt]);
      kf.q[0] = *(const U4*)&KT[cur][nr][32 + hf * 8];      // d-chunk 1 (K=32..63)
      kf.q[1] = *(const U4*)&KT[cur][nr][48 + hf * 8];
      s[nt] = wmma_bf16(qf[1], kf, s[nt]);
    }

    // Online softmax: rows live in 16-lane groups
    float rs[8];
#pragma unroll
    for (int r = 0; r < 8; ++r) {
      float mv = fmaxf(s[0][r], s[1][r]);
#pragma unroll
      for (int off = 1; off < 16; off <<= 1) mv = fmaxf(mv, __shfl_xor(mv, off, 32));
      float mnew = fmaxf(mi[r], mv);
      float p0 = __expf(s[0][r] - mnew);
      float p1 = __expf(s[1][r] - mnew);
      s[0][r] = p0; s[1][r] = p1;
      float sum = p0 + p1;
#pragma unroll
      for (int off = 1; off < 16; off <<= 1) sum += __shfl_xor(sum, off, 32);
      float sc = __expf(mi[r] - mnew);
      li[r] = li[r] * sc + sum;
      mi[r] = mnew;
      rs[r] = sc;
    }
#pragma unroll
    for (int nt = 0; nt < 4; ++nt)
#pragma unroll
      for (int r = 0; r < 8; ++r) acc[nt][r] *= rs[r];

    // P: C-layout -> A-layout via per-wave LDS staging
#pragma unroll
    for (int nt = 0; nt < 2; ++nt)
#pragma unroll
      for (int r = 0; r < 8; ++r)
        PS[wave][hf * 8 + r][nt * 16 + ln] = f2bf(s[nt][r]);
    __syncthreads();

    AB pf;
    pf.q[0] = *(const U4*)&PS[wave][ln][hf * 8];
    pf.q[1] = *(const U4*)&PS[wave][ln][16 + hf * 8];
#pragma unroll
    for (int nt = 0; nt < 4; ++nt) {
      AB vf;
      int nr = nt * 16 + ln;
      vf.q[0] = *(const U4*)&VT[cur][nr][hf * 8];
      vf.q[1] = *(const U4*)&VT[cur][nr][16 + hf * 8];
      acc[nt] = wmma_bf16(pf, vf, acc[nt]);
    }

    wait_async0();       // own async copies for 'nxt' landed
    __syncthreads();     // everyone done reading 'cur'; all copies landed
    cur = nxt;
  }

  // ctx = acc / li ; write in reshape-to-[B,512,64,64] layout for out-proj:
  //   o = h*64 + qq/64 ; p = (qq%64)*64 + d
#pragma unroll
  for (int nt = 0; nt < 4; ++nt) {
#pragma unroll
    for (int r = 0; r < 8; ++r) {
      int qq = q0 + hf * 8 + r;
      int d  = nt * 16 + ln;
      float o = acc[nt][r] / li[r];
      int och = h * 64 + (qq >> 6);
      int p   = ((qq & 63) << 6) + d;
      CTX[((long)b * OC_ + och) * (long)P_ + p] = f2bf(o);
    }
  }
}

// ---------------------------------------------------------------------------
// Host-side orchestration (graph-capture safe: launches only).
// ---------------------------------------------------------------------------
extern "C" void kernel_launch(void* const* d_in, const int* in_sizes, int n_in,
                              void* d_out, int out_size, void* d_ws, size_t ws_size,
                              hipStream_t stream) {
  const float* x      = (const float*)d_in[0];
  const float* qW     = (const float*)d_in[1];
  const float* kW     = (const float*)d_in[2];
  const float* vW     = (const float*)d_in[3];
  const float* oW     = (const float*)d_in[4];
  const float* kdwW   = (const float*)d_in[5];
  const float* kdw_g  = (const float*)d_in[6];
  const float* kdw_b  = (const float*)d_in[7];
  const float* kdw_m  = (const float*)d_in[8];
  const float* kdw_v  = (const float*)d_in[9];
  const float* vdwW   = (const float*)d_in[10];
  const float* vdw_g  = (const float*)d_in[11];
  const float* vdw_b  = (const float*)d_in[12];
  const float* vdw_m  = (const float*)d_in[13];
  const float* vdw_v  = (const float*)d_in[14];

  char* ws = (char*)d_ws;
  size_t off = 0;
  auto take = [&](size_t bytes) -> void* {
    void* p = ws + off;
    off += (bytes + 255) & ~(size_t)255;
    return p;
  };

  __bf16* xbf   = (__bf16*)take((size_t)B_ * C_ * P_ * 2);   // 33.5 MB
  __bf16* qWbf  = (__bf16*)take((size_t)OC_ * C_ * 2);
  __bf16* oWbf  = (__bf16*)take((size_t)C_ * OC_ * 2);
  __bf16* kWef  = (__bf16*)take((size_t)D_ * C_ * 2);
  __bf16* vWef  = (__bf16*)take((size_t)D_ * C_ * 2);
  float*  kbias = (float*)take((size_t)D_ * 4);
  float*  vbias = (float*)take((size_t)D_ * 4);
  __bf16* ykdw  = (__bf16*)take((size_t)B_ * C_ * L_ * 2);   // 8.4 MB
  __bf16* yvdw  = (__bf16*)take((size_t)B_ * C_ * L_ * 2);
  __bf16* Kbf   = (__bf16*)take((size_t)B_ * L_ * D_ * 2);   // 2 MB, [b][l][d]
  __bf16* Vbf   = (__bf16*)take((size_t)B_ * D_ * L_ * 2);   // 2 MB, [b][d][l]
  __bf16* Qbf   = (__bf16*)take((size_t)B_ * NH_ * P_ * D_ * 2);  // 67 MB
  __bf16* CTX   = (__bf16*)take((size_t)B_ * OC_ * P_ * 2);       // 67 MB
  (void)ws_size; (void)n_in; (void)in_sizes; (void)out_size;

  // 1) converts
  {
    long n = (long)B_ * C_ * P_;
    cvt_bf16_kernel<<<dim3((unsigned)((n + 255) / 256)), 256, 0, stream>>>(x, xbf, n);
  }
  cvt_bf16_kernel<<<dim3(512), 256, 0, stream>>>(qW, qWbf, (long)OC_ * C_);
  cvt_bf16_kernel<<<dim3(512), 256, 0, stream>>>(oW, oWbf, (long)C_ * OC_);

  // 2) fold BN into k/v projection weights
  fold_kv_kernel<<<dim3(D_), 256, 0, stream>>>(kW, kdw_g, kdw_b, kdw_m, kdw_v, kWef, kbias);
  fold_kv_kernel<<<dim3(D_), 256, 0, stream>>>(vW, vdw_g, vdw_b, vdw_m, vdw_v, vWef, vbias);

  // 3) depthwise stride-2 convs (k and v in one pass over x)
  dwconv_kernel<<<dim3(L_ / 256, C_, B_), 256, 0, stream>>>(x, kdwW, vdwW, ykdw, yvdw);

  // 4) k / v projections (M=64, N=1024, K=256)
  gemm_bf16_kernel<<<dim3(1, L_ / 64, B_), 256, 0, stream>>>(kWef, ykdw, kbias, Kbf, D_, L_, C_, 1);
  gemm_bf16_kernel<<<dim3(1, L_ / 64, B_), 256, 0, stream>>>(vWef, yvdw, vbias, Vbf, D_, L_, C_, 2);

  // 5) q projection (M=512, N=4096, K=256) with scramble + 1/sqrt(8)
  gemm_bf16_kernel<<<dim3(OC_ / 128, P_ / 64, B_), 256, 0, stream>>>(qWbf, xbf, nullptr, Qbf, OC_, P_, C_, 0);

  // 6) flash attention -> CTX (layout ready for out-proj)
  flash_kernel<<<dim3(P_ / 128, NH_, B_), 256, 0, stream>>>(Qbf, Kbf, Vbf, CTX);

  // 7) output projection (M=256, N=4096, K=512), f32 out
  gemm_bf16_kernel<<<dim3(C_ / 128, P_ / 64, B_), 256, 0, stream>>>(oWbf, CTX, nullptr, (float*)d_out, C_, P_, OC_, 3);
}
